// poly_GIN_adj_79800492359959
// MI455X (gfx1250) — compile-verified, matrix operand order checked by reading, not run on hardware
//
#include <hip/hip_runtime.h>

// ---------------------------------------------------------------------------
// poly_GIN_adj on MI455X (gfx1250, wave32, WMMA)
//
// Roofline: E tensor is 134 MB f32 -> ~5.8us at 23.3 TB/s; the fused
// Em/msg_e GEMM is only 8.6 GFLOP, trivially hidden behind the E stream by
// v_wmma_f32_16x16x32_bf16. All other GEMMs (~1 GFLOP, <5 MB traffic) are
// fused per 16-row tile through LDS. Weights/activations pre-swizzled to
// bf16 fragment order so B-operands load as one 32B global load per lane.
// ---------------------------------------------------------------------------

#define B_  16
#define M_  128
#define H_  256
#define EH_ 128

typedef __attribute__((ext_vector_type(16))) __bf16 v16bf;
typedef __attribute__((ext_vector_type(8)))  __bf16 v8bf;
typedef __attribute__((ext_vector_type(8)))  float  v8f;

// Workspace layout (bytes, all 32B aligned)
#define OFF_MSGE 0u                 // 4 partials * 16*128*128 f32 = 4 MB
#define OFF_WE   4194304u           // 128*128 bf16
#define OFF_WR   4227072u           // 384*256 bf16
#define OFF_W0   4423680u           // 256*256 bf16
#define OFF_W1   4554752u           // 256*256 bf16
#define OFF_XSW  4685824u           // 16*128*256 bf16

__device__ __forceinline__ int kidx(int lane, int e) {
  // 16-bit A/B fragment K index per CDNA5 ISA: elements 0..7 / 8..15 pick K
  // halves; lanes 0-15 vs 16-31 pick sub-halves.
  return e + 8 * ((e >= 8 ? 1 : 0) + (lane >= 16 ? 1 : 0));
}

__device__ __forceinline__ v8f vzero8() {
  v8f z = {0.f, 0.f, 0.f, 0.f, 0.f, 0.f, 0.f, 0.f};
  return z;
}

__device__ __forceinline__ v8f wmma_bf16(v16bf a, v16bf b, v8f c) {
  // D = A(16x32 bf16) * B(32x16 bf16) + C(16x16 f32)
  return __builtin_amdgcn_wmma_f32_16x16x32_bf16(
      false, a, false, b, (short)0, c, false, false);
}

// A-fragment from a row-major bf16 LDS tile (16 rows, padded stride).
// Per lane: two contiguous 16B loads (rows = lane&15; k = 8*(lane>>4) + {0..7, 16..23}).
__device__ __forceinline__ v16bf load_afrag(const __bf16* tile, int stride,
                                            int kt, int lane) {
  int row = lane & 15, hi = (lane >> 4) & 1;
  const __bf16* p = tile + row * stride + kt * 32 + hi * 8;
  union { v16bf v; v8bf h[2]; } u;
  u.h[0] = *(const v8bf*)(p);
  u.h[1] = *(const v8bf*)(p + 16);
  return u.v;
}

// ---------------------------------------------------------------------------
// Prep: swizzle a KxN row-major f32 weight into bf16 fragment order.
// tile = nt*KT + kt ; out[(tile*32 + lane)*16 + e] = W[32kt+kidx, 16nt+lane15]
// ---------------------------------------------------------------------------
__global__ void kswizzleW(const float* __restrict__ W, unsigned short* dstU,
                          int KT, int NT, int N) {
  __bf16* dst = (__bf16*)dstU;
  int tile = blockIdx.x;
  int kt = tile % KT, nt = tile / KT;
  int lane = threadIdx.x;
  int ncol = nt * 16 + (lane & 15);
#pragma unroll
  for (int e = 0; e < 16; ++e) {
    int k = kt * 32 + kidx(lane, e);
    dst[((long)tile * 32 + lane) * 16 + e] = (__bf16)W[(long)k * N + ncol];
  }
}

// Swizzle X[b] (K=i up to 128, N=h up to 256) into per-batch fragment order.
__global__ void kswizzleX(const float* __restrict__ X, unsigned short* dstU) {
  __bf16* dst = (__bf16*)dstU;
  int b = blockIdx.y;
  int tile = blockIdx.x;              // nt*4 + kt, 64 tiles
  int kt = tile & 3, nt = tile >> 2;
  int lane = threadIdx.x;
  int ncol = nt * 16 + (lane & 15);
#pragma unroll
  for (int e = 0; e < 16; ++e) {
    int k = kt * 32 + kidx(lane, e);
    dst[(((long)b * 64 + tile) * 32 + lane) * 16 + e] =
        (__bf16)X[((long)b * M_ + k) * H_ + ncol];
  }
}

// ---------------------------------------------------------------------------
// Kernel A: fused Em = relu(E@We + be) ; msg_e_partial += A[b,i,j] * Em
// Grid: (jtile=8, b=16, isplit=4), 256 threads (8 waves).
// Wave w owns output f-columns [16w, 16w+16).
// A coefficients for the whole 32-i slice are staged once before the loop so
// the HBM-bound inner loop has no divergent branches or scalar side loads.
// ---------------------------------------------------------------------------
__global__ __launch_bounds__(256) void kA_msge(
    const float* __restrict__ E, const float* __restrict__ A,
    const float* __restrict__ be, const unsigned short* WeU,
    float* __restrict__ msge) {
  const __bf16* We_sw = (const __bf16*)WeU;
  int j0 = blockIdx.x << 4;
  int b = blockIdx.y;
  int isplit = blockIdx.z;
  int tid = threadIdx.x, lane = tid & 31, wave = tid >> 5;
  int row = lane & 15, hi = lane >> 4;
  int i0 = isplit * 32;

  __shared__ __align__(16) __bf16 etile[16 * 136];   // row=j, col=e (padded)
  __shared__ float atile[32 * 16];                   // [ii][j]

  // Preload this wave's We fragments (N-tile = wave, 4 K-tiles).
  v16bf wfrag[4];
#pragma unroll
  for (int kt = 0; kt < 4; ++kt)
    wfrag[kt] = *(const v16bf*)(We_sw + (((long)wave * 4 + kt) * 32 + lane) * 16);
  float bias = be[wave * 16 + row];

  // Stage A[b, i0:i0+32, j0:j0+16] once (512 floats, coalesced).
#pragma unroll
  for (int u = 0; u < 2; ++u) {
    int q = u * 256 + tid;                   // q = ii*16 + j
    atile[q] = A[((long)b * M_ + i0 + (q >> 4)) * M_ + j0 + (q & 15)];
  }

  v8f acc = vzero8();
  for (int ii = 0; ii < 32; ++ii) {
    int i = i0 + ii;
    const float* esrc = E + (((long)b * M_ + i) * M_ + j0) * EH_;
    // Stage E[b,i,j0:j0+16,0:128] -> LDS bf16 (2048 floats, float4 loads).
#pragma unroll
    for (int u = 0; u < 2; ++u) {
      int q = u * 256 + tid;                 // 512 float4
      float4 v = *(const float4*)(esrc + (long)q * 4);
      int j = q >> 5, e = (q & 31) << 2;
      __bf16* d = &etile[j * 136 + e];
      d[0] = (__bf16)v.x; d[1] = (__bf16)v.y;
      d[2] = (__bf16)v.z; d[3] = (__bf16)v.w;
    }
    if (ii + 1 < 32)  // speculative prefetch of next E slab -> global_prefetch_b8
      __builtin_prefetch(esrc + (long)M_ * EH_ + tid * 8, 0, 0);
    __syncthreads();

    // Em tile (16 j x 16 f) for this wave: 4 WMMAs over K=128.
    v8f em = vzero8();
#pragma unroll
    for (int kt = 0; kt < 4; ++kt) {
      v16bf a = load_afrag(etile, 136, kt, lane);
      em = wmma_bf16(a, wfrag[kt], em);
    }
    const float* arow = &atile[ii * 16 + 8 * hi];
#pragma unroll
    for (int r = 0; r < 8; ++r) {
      float v = em[r] + bias;
      v = v > 0.f ? v : 0.f;                 // ReLU
      acc[r] += arow[r] * v;                 // A-weighted reduction over i
    }
    __syncthreads();
  }

  float* dst = msge + (long)isplit * B_ * M_ * EH_;
#pragma unroll
  for (int r = 0; r < 8; ++r) {
    int j = j0 + r + 8 * hi;
    dst[((long)b * M_ + j) * EH_ + wave * 16 + row] = acc[r];
  }
}

// ---------------------------------------------------------------------------
// Kernel B: per (b, jtile):
//   msg = [ A^T X , Sum(msg_e partials) ] ; r = (1+eps)X + relu(msg@Wr+br)
//   t1 = relu(r@W0+b0) ; out = relu(t1@W1+b1)
// 256 threads (8 waves); wave w owns N-tiles {2w, 2w+1}.
// ---------------------------------------------------------------------------
__global__ __launch_bounds__(256) void kB_mlp(
    const float* __restrict__ X, const float* __restrict__ A,
    const float* __restrict__ eps, const float* __restrict__ br,
    const float* __restrict__ b0, const float* __restrict__ b1,
    const float* __restrict__ msge, const unsigned short* XswU,
    const unsigned short* WrU, const unsigned short* W0U,
    const unsigned short* W1U, float* __restrict__ out) {
  const __bf16* Xsw   = (const __bf16*)XswU;
  const __bf16* Wr_sw = (const __bf16*)WrU;
  const __bf16* W0_sw = (const __bf16*)W0U;
  const __bf16* W1_sw = (const __bf16*)W1U;

  int wg = blockIdx.x;
  int b = wg >> 3, j0 = (wg & 7) << 4;
  int tid = threadIdx.x, lane = tid & 31, wave = tid >> 5;
  int row = lane & 15, hi = lane >> 4;

  __shared__ __align__(16) __bf16 at[16 * 136];    // A^T tile: row=j, col=i
  __shared__ __align__(16) __bf16 msg[16 * 392];   // row=j, col=c (0..383)
  __shared__ __align__(16) __bf16 t0[16 * 264];
  __shared__ __align__(16) __bf16 t1[16 * 264];

  // Stage A^T tile (transpose during LDS store).
#pragma unroll
  for (int u = 0; u < 8; ++u) {
    int idx = u * 256 + tid;                 // 2048 = 128 i x 16 j
    int i = idx >> 4, j = idx & 15;
    at[j * 136 + i] = (__bf16)A[((long)b * M_ + i) * M_ + j0 + j];
  }
  // Stage msg_e = sum of 4 partials -> msg cols 256..383.
#pragma unroll
  for (int u = 0; u < 2; ++u) {
    int q = u * 256 + tid;                   // 512 float4 = 16 x 128
    int j = q >> 5, e = (q & 31) << 2;
    long base = ((long)b * M_ + j0 + j) * EH_ + e;
    float4 s = *(const float4*)(msge + base);
#pragma unroll
    for (int p = 1; p < 4; ++p) {
      float4 v = *(const float4*)(msge + (long)p * B_ * M_ * EH_ + base);
      s.x += v.x; s.y += v.y; s.z += v.z; s.w += v.w;
    }
    __bf16* d = &msg[j * 392 + 256 + e];
    d[0] = (__bf16)s.x; d[1] = (__bf16)s.y;
    d[2] = (__bf16)s.z; d[3] = (__bf16)s.w;
  }
  __syncthreads();

  // msg_x = A^T(16x128) @ X[b](128x256) via WMMA -> msg cols 0..255.
#pragma unroll
  for (int s = 0; s < 2; ++s) {
    int nt = wave * 2 + s;
    v8f acc = vzero8();
#pragma unroll
    for (int kt = 0; kt < 4; ++kt) {
      v16bf a = load_afrag(at, 136, kt, lane);
      v16bf bf = *(const v16bf*)(Xsw + ((((long)b * 16 + nt) * 4 + kt) * 32 + lane) * 16);
      acc = wmma_bf16(a, bf, acc);
    }
#pragma unroll
    for (int r = 0; r < 8; ++r)
      msg[(r + 8 * hi) * 392 + nt * 16 + row] = (__bf16)acc[r];
  }
  __syncthreads();

  float ep1 = 1.0f + eps[0];
  // Stage R: t0 = (1+eps)X + relu(msg @ Wr + br)     (K=384 -> 12 ktiles)
#pragma unroll
  for (int s = 0; s < 2; ++s) {
    int nt = wave * 2 + s;
    v8f acc = vzero8();
#pragma unroll
    for (int kt = 0; kt < 12; ++kt) {
      v16bf a = load_afrag(msg, 392, kt, lane);
      v16bf bf = *(const v16bf*)(Wr_sw + (((long)nt * 12 + kt) * 32 + lane) * 16);
      acc = wmma_bf16(a, bf, acc);
    }
    float bias = br[nt * 16 + row];
#pragma unroll
    for (int r = 0; r < 8; ++r) {
      float v = acc[r] + bias;
      v = v > 0.f ? v : 0.f;
      float xr = X[((long)b * M_ + j0 + r + 8 * hi) * H_ + nt * 16 + row];
      t0[(r + 8 * hi) * 264 + nt * 16 + row] = (__bf16)(ep1 * xr + v);
    }
  }
  __syncthreads();

  // Stage 0: t1 = relu(t0 @ W0 + b0)                 (K=256 -> 8 ktiles)
#pragma unroll
  for (int s = 0; s < 2; ++s) {
    int nt = wave * 2 + s;
    v8f acc = vzero8();
#pragma unroll
    for (int kt = 0; kt < 8; ++kt) {
      v16bf a = load_afrag(t0, 264, kt, lane);
      v16bf bf = *(const v16bf*)(W0_sw + (((long)nt * 8 + kt) * 32 + lane) * 16);
      acc = wmma_bf16(a, bf, acc);
    }
    float bias = b0[nt * 16 + row];
#pragma unroll
    for (int r = 0; r < 8; ++r) {
      float v = acc[r] + bias;
      t1[(r + 8 * hi) * 264 + nt * 16 + row] = (__bf16)(v > 0.f ? v : 0.f);
    }
  }
  __syncthreads();

  // Stage 1: out = relu(t1 @ W1 + b1) -> global f32.
#pragma unroll
  for (int s = 0; s < 2; ++s) {
    int nt = wave * 2 + s;
    v8f acc = vzero8();
#pragma unroll
    for (int kt = 0; kt < 8; ++kt) {
      v16bf a = load_afrag(t1, 264, kt, lane);
      v16bf bf = *(const v16bf*)(W1_sw + (((long)nt * 8 + kt) * 32 + lane) * 16);
      acc = wmma_bf16(a, bf, acc);
    }
    float bias = b1[nt * 16 + row];
#pragma unroll
    for (int r = 0; r < 8; ++r) {
      float v = acc[r] + bias;
      v = v > 0.f ? v : 0.f;
      out[((long)b * M_ + j0 + r + 8 * hi) * H_ + nt * 16 + row] = v;
    }
  }
}

// ---------------------------------------------------------------------------
extern "C" void kernel_launch(void* const* d_in, const int* in_sizes, int n_in,
                              void* d_out, int out_size, void* d_ws, size_t ws_size,
                              hipStream_t stream) {
  const float* X   = (const float*)d_in[0];
  const float* E   = (const float*)d_in[1];
  const float* A   = (const float*)d_in[2];
  const float* eps = (const float*)d_in[3];
  const float* We  = (const float*)d_in[4];
  const float* be  = (const float*)d_in[5];
  const float* Wr  = (const float*)d_in[6];
  const float* br  = (const float*)d_in[7];
  const float* W0  = (const float*)d_in[8];
  const float* b0  = (const float*)d_in[9];
  const float* W1  = (const float*)d_in[10];
  const float* b1  = (const float*)d_in[11];
  float* out = (float*)d_out;

  char* ws = (char*)d_ws;
  float*          msge  = (float*)(ws + OFF_MSGE);
  unsigned short* We_sw = (unsigned short*)(ws + OFF_WE);
  unsigned short* Wr_sw = (unsigned short*)(ws + OFF_WR);
  unsigned short* W0_sw = (unsigned short*)(ws + OFF_W0);
  unsigned short* W1_sw = (unsigned short*)(ws + OFF_W1);
  unsigned short* X_sw  = (unsigned short*)(ws + OFF_XSW);

  // Prep: bf16 fragment-order swizzles (tiny; one wave per 32x16 tile).
  kswizzleW<<<32,  32, 0, stream>>>(We, We_sw, 4, 8, 128);
  kswizzleW<<<192, 32, 0, stream>>>(Wr, Wr_sw, 12, 16, 256);
  kswizzleW<<<128, 32, 0, stream>>>(W0, W0_sw, 8, 16, 256);
  kswizzleW<<<128, 32, 0, stream>>>(W1, W1_sw, 8, 16, 256);
  kswizzleX<<<dim3(64, 16), 32, 0, stream>>>(X, X_sw);

  // Dominant kernel: stream 134 MB of E once, fused Em + A-reduction.
  kA_msge<<<dim3(8, 16, 4), 256, 0, stream>>>(E, A, be, We_sw, msge);

  // Fused msg_x + 3-stage MLP chain.
  kB_mlp<<<128, 256, 0, stream>>>(X, A, eps, br, b0, b1, msge, X_sw,
                                  Wr_sw, W0_sw, W1_sw, out);
}
